// NNUE_15358803050934
// MI455X (gfx1250) — matrix operand order repaired and checked by reading, not compiled, not run on hardware
//
#include <hip/hip_runtime.h>

#define B_TOTAL 16384
#define BAG     32
#define L1      512
#define PAD     768
#define NH      8
#define ROWS    16     // batch rows per block -> WMMA M dimension
#define THREADS 256    // 8 wave32

typedef __attribute__((ext_vector_type(16))) _Float16 v16h;
typedef __attribute__((ext_vector_type(8)))  float    v8f;

union HFrag { unsigned int u[8]; v16h v; };

__global__ __launch_bounds__(THREADS)
void nnue_fused_kernel(const int*   __restrict__ stm_idx,
                       const int*   __restrict__ nstm_idx,
                       const float* __restrict__ emb_table,
                       const float* __restrict__ emb_bias,
                       const float* __restrict__ head_w,
                       const float* __restrict__ head_b,
                       float*       __restrict__ out)
{
    __shared__ __align__(16) _Float16 emb_lds[ROWS * 2 * L1];      // 16 x 1024 fp16 = 32 KB
    __shared__ __align__(16) _Float16 hw_lds[(NH + 1) * 2 * L1];   // 9 x 1024 fp16 = 18 KB (row 8 = zeros)
    __shared__ __align__(16) float    part_lds[8 * 16 * 16];       // 8 waves x 16x16 f32 = 8 KB
    __shared__ int bucket_lds[ROWS];

    const int tid  = threadIdx.x;
    const int row  = tid >> 4;       // 0..15 : batch row within tile
    const int sub  = tid & 15;       // 16 threads cooperate on one row
    const int gRow = blockIdx.x * ROWS + row;
    const int c0   = sub * 32;       // 32 contiguous L1 columns per thread

    // ---- stage head weights to LDS fp16, layout [n][k], extra zero row for N-padding ----
    for (int e = tid; e < (NH + 1) * 2 * L1; e += THREADS) {
        const int n = e >> 10, k = e & (2 * L1 - 1);
        hw_lds[e] = (n < NH) ? (_Float16)head_w[n * (2 * L1) + k] : (_Float16)0;
    }

    const float4* tbl4 = (const float4*)emb_table;
    float acc[32];

    // ---- STM bag: accumulate (bias as init), count non-pad for bucket ----
    #pragma unroll
    for (int k = 0; k < 32; ++k) acc[k] = emb_bias[c0 + k];
    int cnt = 0;
    for (int i = 0; i < BAG; ++i) {
        const int idx = stm_idx[gRow * BAG + i];
        if (i + 1 < BAG) {
            const int nidx = stm_idx[gRow * BAG + i + 1];
            __builtin_prefetch(emb_table + (size_t)nidx * L1 + c0, 0, 1);
        }
        cnt += (idx != PAD);
        if (idx != PAD) {
            const float4* src = tbl4 + (((size_t)idx * L1 + c0) >> 2);
            #pragma unroll
            for (int q = 0; q < 8; ++q) {
                const float4 v = src[q];
                acc[4*q+0] += v.x; acc[4*q+1] += v.y;
                acc[4*q+2] += v.z; acc[4*q+3] += v.w;
            }
        }
    }
    #pragma unroll
    for (int k = 0; k < 32; ++k) {
        float v = fminf(fmaxf(acc[k], 0.f), 1.f);          // clip [0,1]
        emb_lds[row * (2 * L1) + c0 + k] = (_Float16)v;
    }
    if (sub == 0) {
        int b = (cnt - 2) / 4;
        b = b < 0 ? 0 : (b > 7 ? 7 : b);
        bucket_lds[row] = b;
    }

    // ---- NSTM bag ----
    #pragma unroll
    for (int k = 0; k < 32; ++k) acc[k] = emb_bias[c0 + k];
    for (int i = 0; i < BAG; ++i) {
        const int idx = nstm_idx[gRow * BAG + i];
        if (i + 1 < BAG) {
            const int nidx = nstm_idx[gRow * BAG + i + 1];
            __builtin_prefetch(emb_table + (size_t)nidx * L1 + c0, 0, 1);
        }
        if (idx != PAD) {
            const float4* src = tbl4 + (((size_t)idx * L1 + c0) >> 2);
            #pragma unroll
            for (int q = 0; q < 8; ++q) {
                const float4 v = src[q];
                acc[4*q+0] += v.x; acc[4*q+1] += v.y;
                acc[4*q+2] += v.z; acc[4*q+3] += v.w;
            }
        }
    }
    #pragma unroll
    for (int k = 0; k < 32; ++k) {
        float v = fminf(fmaxf(acc[k], 0.f), 1.f);
        emb_lds[row * (2 * L1) + L1 + c0 + k] = (_Float16)v;
    }

    __syncthreads();

    // ---- head matvec via WMMA: [16 x 1024] fp16 x [1024 x 16] fp16 -> [16 x 16] f32 ----
    const int wave   = tid >> 5;     // 0..7, each takes 4 of the 32 K-chunks
    const int lane   = tid & 31;
    const int laneLo = lane & 15;
    const int laneHi = lane >> 4;
    const int nRow   = (laneLo < NH) ? laneLo : NH;   // branch-free zero-pad column select

    v8f c = {0.f, 0.f, 0.f, 0.f, 0.f, 0.f, 0.f, 0.f};
    #pragma unroll
    for (int t = 0; t < 4; ++t) {
        const int kOff = (wave + t * 8) * 32;
        HFrag a, b;
        #pragma unroll
        for (int v = 0; v < 8; ++v) {
            // A (16x32 f16): VGPR v holds K pair; lanes 16-31 shift K by +8
            const int KbA = 2 * (v & 3) + ((v >= 4) ? 16 : 0) + (laneHi ? 8 : 0);
            a.u[v] = *(const unsigned int*)&emb_lds[laneLo * (2 * L1) + kOff + KbA];
            // B (32x16 f16): same K order as A; lanes 16-31 shift K by +8
            const int KbB = 2 * (v & 3) + ((v >= 4) ? 16 : 0) + (laneHi ? 8 : 0);
            b.u[v] = *(const unsigned int*)&hw_lds[nRow * (2 * L1) + kOff + KbB];
        }
        c = __builtin_amdgcn_wmma_f32_16x16x32_f16(
                /*neg_a=*/false, a.v, /*neg_b=*/false, b.v,
                /*c_mod=*/(short)0, c, /*reuse_a=*/false, /*reuse_b=*/false);
    }
    // C layout: VGPR j -> M = j (+8 for lanes 16..31), N = lane-lo
    #pragma unroll
    for (int j = 0; j < 8; ++j) {
        const int m = j + (laneHi ? 8 : 0);
        part_lds[(wave * 16 + m) * 16 + laneLo] = c[j];
    }

    __syncthreads();

    // ---- cross-wave reduction + bucket select ----
    {
        const int m = tid >> 4, n = tid & 15;
        if (n < NH) {
            float s = head_b[n];
            #pragma unroll
            for (int w = 0; w < 8; ++w) s += part_lds[(w * 16 + m) * 16 + n];
            if (n == bucket_lds[m]) out[blockIdx.x * ROWS + m] = s;
        }
    }
}

extern "C" void kernel_launch(void* const* d_in, const int* in_sizes, int n_in,
                              void* d_out, int out_size, void* d_ws, size_t ws_size,
                              hipStream_t stream) {
    (void)in_sizes; (void)n_in; (void)out_size; (void)d_ws; (void)ws_size;
    const int*   stm  = (const int*)  d_in[0];
    const int*   nstm = (const int*)  d_in[1];
    const float* tbl  = (const float*)d_in[2];
    const float* bias = (const float*)d_in[3];
    const float* hw   = (const float*)d_in[4];
    const float* hb   = (const float*)d_in[5];
    float* out = (float*)d_out;

    dim3 grid(B_TOTAL / ROWS);   // 1024 blocks x 16 rows = 16384
    dim3 block(THREADS);
    nnue_fused_kernel<<<grid, block, 0, stream>>>(stm, nstm, tbl, bias, hw, hb, out);
}
